// HydraMultivariate_89670327206026
// MI455X (gfx1250) — compile-verified, matrix operand order checked by reading, not run on hardware
//
#include <hip/hip_runtime.h>

typedef __attribute__((ext_vector_type(16))) _Float16 v16h;
typedef __attribute__((ext_vector_type(8)))  _Float16 v8h;
typedef __attribute__((ext_vector_type(8)))  float    v8f;
typedef __attribute__((ext_vector_type(4)))  float    v4f;

#define B_    128
#define C_    12
#define L_    1024
#define NDIL  7
#define DIV_  2
#define H_    32
#define K_    8
#define KSZ   9
#define NCP   6
#define RED   (C_*KSZ)            // 108 (channels x taps)
#define REDP  128                 // padded reduction: 4x K=32 WMMA steps
#define FEAT  (NDIL*DIV_*2*H_*K_) // 7168

#define PAD   256                 // max tap reach: 4 * 2^6
#define CROW  (PAD + L_ + PAD)    // 1536 halves per channel in LDS
#define ZOFF  (C_ * CROW)         // start of the always-zero scratch row
#define SXN   (ZOFF + 1024)       // total staged halves = 19456 (38 KiB)

// (s,i) pairs whose reduction index is >= RED for BOTH lane halves (hi=0/1):
// r_min = 32*s + (i&7) + 16*(i>>3) >= 108  <=>  s==3 && i>=8.
// Those B halves are compile-time zeros: no LDS load, no address math.
#define ALL_INVALID(s, i) ((32*(s) + ((i)&7) + 16*((i)>>3)) >= RED)

// ---------------------------------------------------------------------------
// Fold the irregular channel gather into the conv weights:
//   Weff[z][oc][c*9+t] = W[z][oc][t] * (# of times channel c appears in
//   I[z][oc/8]).   f16, reduction padded 108 -> 128.
// ---------------------------------------------------------------------------
__global__ void hydra_weff_kernel(const float* __restrict__ W,
                                  const int*   __restrict__ I,
                                  _Float16*    __restrict__ weff) {
    int z  = blockIdx.x;        // di*2 + j, 0..13
    int oc = threadIdx.x;       // 0..255
    int h  = oc >> 3;
    const int* Ih = I + (z * H_ + h) * NCP;
    int i0 = Ih[0], i1 = Ih[1], i2 = Ih[2], i3 = Ih[3], i4 = Ih[4], i5 = Ih[5];
    const float* w = W + ((size_t)z * (K_ * H_) + oc) * KSZ;
    float wt[KSZ];
#pragma unroll
    for (int t = 0; t < KSZ; ++t) wt[t] = w[t];
    _Float16* o = weff + ((size_t)z * 256 + oc) * REDP;
#pragma unroll
    for (int c = 0; c < C_; ++c) {
        float cnt = (float)((i0==c)+(i1==c)+(i2==c)+(i3==c)+(i4==c)+(i5==c));
#pragma unroll
        for (int t = 0; t < KSZ; ++t)
            o[c * KSZ + t] = (_Float16)(wt[t] * cnt);
    }
#pragma unroll
    for (int r = RED; r < REDP; ++r) o[r] = (_Float16)0.0f;
}

// ---------------------------------------------------------------------------
// Main kernel: one 512-thread block (16 wave32s) per (batch b, z = di*2+j).
// The 12x1024 series is staged ONCE into padded LDS (f16, diff-on-the-fly for
// j==1); wave w computes head pair (2w, 2w+1):
//   D[16 oc x 16 l] = A[16 oc x 128] * B[128 x 16 l]  via 4x
//   v_wmma_f32_16x16x32_f16 per l-tile.  Lane n holds the full K=8 group of
//   one head at l = l0+n in its 8 f32 accumulators -> per-lane argmax/argmin,
//   wave-private LDS histograms, plain final store (no global atomics).
// ---------------------------------------------------------------------------
__global__ void __launch_bounds__(512)
hydra_wmma_kernel(const float*    __restrict__ X,
                  const _Float16* __restrict__ weff,
                  float*          __restrict__ out) {
    __shared__ _Float16 s_x[SXN];       // padded staged series + zero row
    __shared__ float    s_cnt[16 * 32]; // per-wave [type(2)][head(2)][k(8)]

    const int tid  = threadIdx.x;
    const int lane = tid & 31;
    const int w    = tid >> 5;          // wave id == head-pair tile (0..15)
    const int b    = blockIdx.x;        // 0..127
    const int z    = blockIdx.y;        // 0..13  (di*2 + j)
    const int di   = z >> 1;
    const int j    = z & 1;
    const int d    = 1 << di;
    const int Lsrc = (j == 0) ? L_ : (L_ - 1);

    const float* src = X + (size_t)b * C_ * L_;

    // ---- zero all staged LDS (pads + zero row), then overlay the data ----
    {
        unsigned long long* z64 = reinterpret_cast<unsigned long long*>(s_x);
        for (int i = tid; i < SXN / 4; i += 512) z64[i] = 0ull;
    }
    s_cnt[tid] = 0.0f;
    __syncthreads();

    // ---- stage 12x1024 f32 -> padded f16 rows (diff-on-the-fly for j==1) ---
#pragma unroll
    for (int it = 0; it < (C_ * L_) / (512 * 4); ++it) {   // 6 iterations
        int q = it * 512 + tid;
        int e = q * 4;                       // flat f32 element index
        int c = e >> 10;
        int l = e & (L_ - 1);
        v4f v = *reinterpret_cast<const v4f*>(src + e);
        _Float16 h0, h1, h2, h3;
        if (j == 0) {
            h0 = (_Float16)v.x; h1 = (_Float16)v.y;
            h2 = (_Float16)v.z; h3 = (_Float16)v.w;
        } else {
            bool last = (l == L_ - 4);
            float nxt = last ? 0.0f : src[e + 4];
            h0 = (_Float16)(v.y - v.x);
            h1 = (_Float16)(v.z - v.y);
            h2 = (_Float16)(v.w - v.z);
            h3 = last ? (_Float16)0.0f : (_Float16)(nxt - v.w);
        }
        union { _Float16 h[4]; unsigned long long u; } pk4;
        pk4.h[0] = h0; pk4.h[1] = h1; pk4.h[2] = h2; pk4.h[3] = h3;
        *reinterpret_cast<unsigned long long*>(&s_x[c * CROW + PAD + l]) = pk4.u;
    }
    __syncthreads();

    const int m  = lane & 15;   // A row / B,D column within tile
    const int hi = lane >> 4;   // K-half selector per ISA WMMA VGPR layout

    // ---- load A (16x128 Weff tile) once; lane m holds row oc = w*16 + m.
    // half i of step s holds red index r = 32*s + (i&7) + 8*hi + 16*(i>>3)
    const _Float16* wrow = weff + ((size_t)z * 256 + w * 16 + m) * REDP;
    v16h a[4];
#pragma unroll
    for (int s = 0; s < 4; ++s) {
        union { v8h p[2]; v16h full; } ua;
        ua.p[0] = *reinterpret_cast<const v8h*>(wrow + 32 * s + 8 * hi);
        ua.p[1] = *reinterpret_cast<const v8h*>(wrow + 32 * s + 16 + 8 * hi);
        a[s] = ua.full;
    }

    // ---- per-lane base LDS indices for the B-gather taps (branchless:
    // per-lane-invalid/pad taps hit the zero row / zero pads; taps invalid
    // for ALL lanes are folded to literal zero and never touch LDS)
    int pb[4][16];
#pragma unroll
    for (int s = 0; s < 4; ++s) {
#pragma unroll
        for (int i = 0; i < 16; ++i) {
            if (ALL_INVALID(s, i)) { pb[s][i] = 0; continue; }
            int k = (i & 7) + 8 * hi + 16 * (i >> 3);
            int r = 32 * s + k;
            if (r < RED) {
                int c = r / KSZ;
                int t = r - KSZ * c;
                pb[s][i] = c * CROW + PAD + m + (t - 4) * d;
            } else {
                pb[s][i] = ZOFF;            // reads 0 for any l0
            }
        }
    }

    // ---- main loop: 64 l-tiles of 16 positions; unroll 4 so the uniform
    // +l0 mostly folds into ds_load immediate offsets ----
#pragma unroll 4
    for (int l0 = 0; l0 < L_; l0 += 16) {
        v8f acc = {};
#pragma unroll
        for (int s = 0; s < 4; ++s) {
            union { _Float16 h[16]; v16h v; } bu;
#pragma unroll
            for (int i = 0; i < 16; ++i) {
                if (ALL_INVALID(s, i)) bu.h[i] = (_Float16)0.0f;
                else                   bu.h[i] = s_x[pb[s][i] + l0];
            }
            acc = __builtin_amdgcn_wmma_f32_16x16x32_f16(
                      false, a[s], false, bu.v, (short)0, acc, false, false);
        }
        // epilogue: this lane owns y[head hi][k=0..7] at l = l0 + m
        float mx = acc[0], mn = acc[0];
        int amx = 0, amn = 0;
#pragma unroll
        for (int k = 1; k < 8; ++k) {
            float v = acc[k];
            if (v > mx) { mx = v; amx = k; }   // first max (argmax semantics)
            if (v < mn) { mn = v; amn = k; }   // first min
        }
        int l = l0 + m;
        if (l < Lsrc) {
            atomicAdd(&s_cnt[w * 32 + hi * 8 + amx], mx);
            atomicAdd(&s_cnt[w * 32 + 16 + hi * 8 + amn], 1.0f);
        }
    }

    // ---- exclusive output slice (wave-private histograms): plain stores ----
    int type = lane >> 4;          // 0 = count_max, 1 = count_min
    int hh   = (lane >> 3) & 1;    // which head of the pair
    int kk   = lane & 7;
    out[(size_t)b * FEAT + (z * 2 + type) * 256 + (w * 2 + hh) * 8 + kk] =
        s_cnt[w * 32 + type * 16 + hh * 8 + kk];
}

// ---------------------------------------------------------------------------
extern "C" void kernel_launch(void* const* d_in, const int* in_sizes, int n_in,
                              void* d_out, int out_size, void* d_ws, size_t ws_size,
                              hipStream_t stream) {
    const float* X = (const float*)d_in[0];   // [128,12,1024] f32
    const float* W = (const float*)d_in[1];   // [7,2,256,1,9] f32
    const int*   I = (const int*)d_in[2];     // [7,2,32,6] i32
    float* out     = (float*)d_out;           // [128,7168] f32
    _Float16* weff = (_Float16*)d_ws;         // 14*256*128 f16 = 896 KiB

    hydra_weff_kernel<<<dim3(NDIL * DIV_), dim3(256), 0, stream>>>(W, I, weff);
    hydra_wmma_kernel<<<dim3(B_, NDIL * DIV_), dim3(512), 0, stream>>>(
        X, weff, out);
}
